// Traffic_prediction_model_35665408426660
// MI455X (gfx1250) — matrix operand order repaired and checked by reading, not compiled, not run on hardware
//
#include <hip/hip_runtime.h>

// ---------------- problem constants ----------------
#define Bb    16
#define Nn    512      // conv input channels
#define Ff    8
#define Tt    1024
#define Hh    256
#define Ee    16384
#define TNv   1022
#define OC    512      // 2*H conv output channels
#define KDIM  1536     // N*3 im2col reduction
#define KT    32       // K tile (one wmma K)
#define NT    32       // t-column tile
#define BF    128      // B*F conv batches
#define KSTEPS (KDIM / KT)   // 48
#define STAGE_BYTES 34816    // 32KB A + 2KB B per stage

typedef __attribute__((ext_vector_type(16))) __bf16 v16bf;
typedef __attribute__((ext_vector_type(8)))  float  v8f;

union FragAB { v16bf v; uint4 u[2]; };

// gfx1250 async global->LDS 16B copy (ASYNCcnt-tracked, no VGPR staging)
__device__ __forceinline__ void async_copy16(const __bf16* gptr, __bf16* lptr) {
  unsigned lds = (unsigned)(uintptr_t)lptr;   // low 32 bits of generic ptr = LDS offset
  unsigned long long ga = (unsigned long long)(uintptr_t)gptr;
  asm volatile("global_load_async_to_lds_b128 %0, %1, off"
               :: "v"(lds), "v"(ga) : "memory");
}
__device__ __forceinline__ void wait_async0() {
  asm volatile("s_wait_asynccnt 0x0" ::: "memory");
}

// ---------------- fp32 -> bf16 (vector x4) ----------------
__global__ __launch_bounds__(256)
void f32_to_bf16_kernel(const float* __restrict__ in, __bf16* __restrict__ out, int n4) {
  int i = blockIdx.x * blockDim.x + threadIdx.x;
  if (i >= n4) return;
  float4 v = ((const float4*)in)[i];
  union { __bf16 h[4]; uint2 u; } p;
  p.h[0] = (__bf16)v.x; p.h[1] = (__bf16)v.y;
  p.h[2] = (__bf16)v.z; p.h[3] = (__bf16)v.w;
  ((uint2*)out)[i] = p.u;
}

// ---------------- fused conv1d-as-GEMM (WMMA bf16) + GLU ----------------
// grid = (32 t-tiles, 128 bf), block = 256 (8 waves), dynamic LDS = 69632 B.
// Double-buffered pipeline: async A-tile (weights) global->LDS overlapped with
// WMMA on current stage; B (im2col) gathered to regs pre-compute, stored post.
__global__ __launch_bounds__(256)
void conv_glu_kernel(const __bf16* __restrict__ xbf,   // [128][512][1024]
                     const __bf16* __restrict__ wbf,   // [512][1536]
                     const float*  __restrict__ convb, // [512]
                     float*        __restrict__ gbuf)  // [B*H][1022][8]
{
  extern __shared__ char smem[];

  const int tid  = threadIdx.x;
  const int wave = tid >> 5;
  const int lane = tid & 31;
  const int hl   = lane >> 4;   // half-wave select
  const int ln   = lane & 15;

  const int  t0    = blockIdx.x * NT;       // t-column base
  const int  bf    = blockIdx.y;            // 0..127
  const long xbase = (long)bf * Nn * Tt;

  v8f acc[4][2];
  for (int mf = 0; mf < 4; ++mf)
    for (int nf = 0; nf < 2; ++nf)
      acc[mf][nf] = v8f{};

  auto Abuf = [&](int s) { return (__bf16*)(smem + s * STAGE_BYTES); };
  auto Bbuf = [&](int s) { return (__bf16*)(smem + s * STAGE_BYTES + OC * KT * 2); };

  // A tile: 512x32 weights, 4 x 16B async copies per thread
  auto issueA = [&](int k0, int s) {
    __bf16* ab = Abuf(s);
    for (int j = 0; j < 4; ++j) {
      int i   = tid + j * 256;
      int row = i >> 1, seg = i & 1;
      async_copy16(wbf + (long)row * KDIM + k0 + seg * 16,
                   ab + row * KT + seg * 16);
    }
  };
  // B tile: im2col gather, 4 bf16 per thread held in regs
  auto loadB = [&](int k0, __bf16* v) {
    for (int j = 0; j < 4; ++j) {
      int i  = tid + j * 256;
      int kr = i & (KT - 1);
      int tr = i >> 5;
      int kk = k0 + kr;
      int ic = kk / 3;
      int ko = kk - ic * 3;
      int t  = t0 + tr;
      v[j] = (t < TNv) ? xbf[xbase + (long)ic * Tt + t + ko] : (__bf16)0.f;
    }
  };
  auto storeB = [&](int s, const __bf16* v) {
    __bf16* bb = Bbuf(s);
    for (int j = 0; j < 4; ++j) {
      int i  = tid + j * 256;
      int kr = i & (KT - 1);
      int tr = i >> 5;
      bb[tr * KT + kr] = v[j];
    }
  };

  // prologue: stage 0
  __bf16 bstage[4];
  issueA(0, 0);
  loadB(0, bstage);
  storeB(0, bstage);

  for (int k = 0; k < KSTEPS; ++k) {
    const int cur = k & 1, nxt = cur ^ 1;
    wait_async0();        // our async A(k) copies landed in LDS
    __syncthreads();      // everyone's stage-k data visible; stage-(k-1) reads done

    if (k + 1 < KSTEPS) { // start streaming stage k+1 under the WMMAs
      issueA((k + 1) * KT, nxt);
      loadB((k + 1) * KT, bstage);
    }

    // B fragments: lanes 0-15 -> col n, K{0..7,16..23} ; lanes 16-31 -> K{8..15,24..31}
    const __bf16* al = Abuf(cur);
    const __bf16* bl = Bbuf(cur);
    FragAB bfr[2], afr[4];
    for (int nf = 0; nf < 2; ++nf) {
      const __bf16* bp = bl + (nf * 16 + ln) * KT + hl * 8;
      bfr[nf].u[0] = *(const uint4*)(bp);
      bfr[nf].u[1] = *(const uint4*)(bp + 16);
    }
    // A fragments: lanes 0-15 -> K{0..7,16..23} ; lanes 16-31 -> K{8..15,24..31}
    for (int mf = 0; mf < 4; ++mf) {
      const __bf16* ap = al + (wave * 64 + mf * 16 + ln) * KT;
      afr[mf].u[0] = *(const uint4*)(ap + hl * 8);
      afr[mf].u[1] = *(const uint4*)(ap + 16 + hl * 8);
    }
    for (int mf = 0; mf < 4; ++mf)
      for (int nf = 0; nf < 2; ++nf)
        acc[mf][nf] = __builtin_amdgcn_wmma_f32_16x16x32_bf16(
            false, afr[mf].v, false, bfr[nf].v, (short)0, acc[mf][nf], false, false);

    if (k + 1 < KSTEPS) storeB(nxt, bstage);  // land B(k+1) after compute issues
  }

  // spill accumulators to LDS (C/D layout: vgpr j -> row m0+hl*8+j, col n0+ln)
  __syncthreads();
  float* Clds = (float*)smem;   // 512*32 f32 = 64 KB, reuses both stages
  for (int mf = 0; mf < 4; ++mf)
    for (int nf = 0; nf < 2; ++nf) {
      int m0 = wave * 64 + mf * 16 + hl * 8;
      int n0 = nf * 16 + ln;
      for (int j = 0; j < 8; ++j)
        Clds[(m0 + j) * NT + n0] = acc[mf][nf][j];
    }
  __syncthreads();

  // GLU epilogue: g = (P+b) * sigmoid(Q+b); write in [b][h][t][f] layout
  const int b  = bf >> 3, f = bf & 7;
  const int h  = tid; // 256 threads == H rows
  const float pb = convb[h];
  const float qb = convb[h + Hh];
  for (int c = 0; c < NT; ++c) {
    int t = t0 + c;
    if (t >= TNv) break;
    float P = Clds[h * NT + c] + pb;
    float Q = Clds[(h + Hh) * NT + c] + qb;
    float g = P * (1.f / (1.f + __expf(-Q)));
    gbuf[(((long)(b * Hh + h)) * TNv + t) * Ff + f] = g;
  }
}

// ---------------- in-place hw = g @ gcn_w (8x8) ----------------
__global__ __launch_bounds__(256)
void gcn_hw_kernel(float* __restrict__ gbuf, const float* __restrict__ gw, int nrows) {
  __shared__ float Wsh[64];
  if (threadIdx.x < 64) Wsh[threadIdx.x] = gw[threadIdx.x];
  __syncthreads();
  int r = blockIdx.x * blockDim.x + threadIdx.x;
  if (r >= nrows) return;
  float* p = gbuf + (long)r * 8;
  float4 lo = *(float4*)p, hi = *(float4*)(p + 4);
  float in[8] = {lo.x, lo.y, lo.z, lo.w, hi.x, hi.y, hi.z, hi.w};
  float o[8];
  for (int g = 0; g < 8; ++g) {
    float s = 0.f;
    for (int f = 0; f < 8; ++f) s += in[f] * Wsh[f * 8 + g];
    o[g] = s;
  }
  float4 olo = {o[0], o[1], o[2], o[3]};
  float4 ohi = {o[4], o[5], o[6], o[7]};
  *(float4*)p = olo;
  *(float4*)(p + 4) = ohi;
}

// ---------------- degree ----------------
__global__ void deg_init_kernel(float* __restrict__ deg) {
  int i = blockIdx.x * blockDim.x + threadIdx.x;
  if (i < TNv) deg[i] = 1.f; // self loop
}
__global__ void deg_edge_kernel(const int* __restrict__ ei, float* __restrict__ deg) {
  int e = blockIdx.x * blockDim.x + threadIdx.x;
  if (e >= Ee) return;
  int d = ei[Ee + e];
  if ((unsigned)d < (unsigned)TNv) atomicAdd(deg + d, 1.f);
}

// ---------------- self-loop + bias init, writes d_out in (B,F,TN,H) ----------------
__global__ __launch_bounds__(256)
void gcn_init_kernel(const float* __restrict__ hwb, const float* __restrict__ deg,
                     const float* __restrict__ gb, float* __restrict__ out, int n) {
  int i = blockIdx.x * blockDim.x + threadIdx.x;
  if (i >= n) return;
  int h   = i & 255;
  int t   = (i >> 8) % TNv;
  int rem = (i >> 8) / TNv;
  int f   = rem & 7;
  int b   = rem >> 3;
  float di = rsqrtf(deg[t]);
  float v  = hwb[(((long)(b * Hh + h)) * TNv + t) * Ff + f];
  out[i] = v * di * di + gb[f];
}

// ---------------- edge scatter-add into d_out ----------------
__global__ __launch_bounds__(256)
void gcn_scatter_kernel(const float* __restrict__ hwb, const float* __restrict__ deg,
                        const int* __restrict__ ei, float* __restrict__ out) {
  long id = (long)blockIdx.x * blockDim.x + threadIdx.x; // 4096 * 16384 threads
  int e  = (int)(id & (Ee - 1));
  int bh = (int)(id >> 14);
  int s = ei[e];
  int d = ei[Ee + e];
  if ((unsigned)s >= (unsigned)TNv || (unsigned)d >= (unsigned)TNv) return;
  float norm = rsqrtf(deg[s]) * rsqrtf(deg[d]);
  const float* src = hwb + (((long)bh) * TNv + s) * Ff;
  int b = bh >> 8, h = bh & 255;
  long base = ((long)(b * Ff) * TNv + d) * Hh + h;
  for (int f = 0; f < Ff; ++f)
    atomicAdd(out + base + (long)f * TNv * Hh, src[f] * norm);
}

// ---------------- launch ----------------
extern "C" void kernel_launch(void* const* d_in, const int* in_sizes, int n_in,
                              void* d_out, int out_size, void* d_ws, size_t ws_size,
                              hipStream_t stream) {
  const float* x      = (const float*)d_in[0];
  const int*   eidx   = (const int*)d_in[1];
  const float* conv_w = (const float*)d_in[3];
  const float* conv_b = (const float*)d_in[4];
  const float* gcn_w  = (const float*)d_in[5];
  const float* gcn_b  = (const float*)d_in[6];
  float* out = (float*)d_out;

  // workspace carving (256B aligned)
  char* ws = (char*)d_ws;
  const size_t xbf_bytes = (size_t)BF * Nn * Tt * 2;          // 134,217,728
  const size_t wbf_bytes = (size_t)OC * KDIM * 2;             //   1,572,864
  const size_t g_bytes   = (size_t)Bb * Hh * TNv * Ff * 4;    // 133,955,584
  __bf16* xbf  = (__bf16*)ws;
  __bf16* wbf  = (__bf16*)(ws + xbf_bytes);
  float*  gbuf = (float*)(ws + xbf_bytes + wbf_bytes);
  float*  deg  = (float*)(ws + xbf_bytes + wbf_bytes + g_bytes);

  // 1) convert inputs to bf16
  {
    int n4 = (BF * Nn * Tt) / 4; // 16,777,216
    f32_to_bf16_kernel<<<(n4 + 255) / 256, 256, 0, stream>>>(x, xbf, n4);
    int w4 = (OC * KDIM) / 4;    //    196,608
    f32_to_bf16_kernel<<<(w4 + 255) / 256, 256, 0, stream>>>(conv_w, wbf, w4);
  }
  // 2) fused conv-GEMM (WMMA, async double-buffered) + GLU
  {
    dim3 grid(32, BF, 1);
    conv_glu_kernel<<<grid, 256, 2 * STAGE_BYTES, stream>>>(xbf, wbf, conv_b, gbuf);
  }
  // 3) in-place 8x8 feature transform
  {
    int nrows = Bb * Hh * TNv; // 4,186,112
    gcn_hw_kernel<<<(nrows + 255) / 256, 256, 0, stream>>>(gbuf, gcn_w, nrows);
  }
  // 4) degrees
  deg_init_kernel<<<(TNv + 255) / 256, 256, 0, stream>>>(deg);
  deg_edge_kernel<<<(Ee + 255) / 256, 256, 0, stream>>>(eidx, deg);
  // 5) self-loop + bias into d_out (final layout)
  {
    int n = Bb * Ff * TNv * Hh; // 33,488,896
    gcn_init_kernel<<<(n + 255) / 256, 256, 0, stream>>>(gbuf, deg, gcn_b, out, n);
  }
  // 6) edge scatter-add
  {
    long total = (long)Bb * Hh * Ee; // 67,108,864
    gcn_scatter_kernel<<<(int)(total / 256), 256, 0, stream>>>(gbuf, deg, eidx, out);
  }
}
